// GPT2Block_39917426049328
// MI455X (gfx1250) — matrix operand order repaired and checked by reading, not compiled, not run on hardware
//
#include <hip/hip_runtime.h>
#include <math.h>

// ---------------- CDNA5 WMMA types ----------------
typedef __attribute__((ext_vector_type(16))) _Float16 v16h;
typedef __attribute__((ext_vector_type(8)))  _Float16 v8h;
typedef __attribute__((ext_vector_type(8)))  float    v8f;

#define WMMA_F32_F16(a, b, c) \
  __builtin_amdgcn_wmma_f32_16x16x32_f16(false, (a), false, (b), (short)0, (c), false, false)

__device__ __forceinline__ v16h cat16(v8h lo, v8h hi) {
  return __builtin_shufflevector(lo, hi, 0, 1, 2, 3, 4, 5, 6, 7,
                                 8, 9, 10, 11, 12, 13, 14, 15);
}

// ---------------- problem constants ----------------
constexpr int Bb  = 2;
constexpr int Ss  = 2048;
constexpr int Dd  = 1024;
constexpr int Hh  = 16;
constexpr int HDd = 64;        // head dim
constexpr int Mrows = Bb * Ss; // 4096

// ---------------- helpers ----------------
struct alignas(16) H8 { _Float16 h[8]; };
struct alignas(4)  H2 { _Float16 x, y; };

__device__ __forceinline__ float gelu_new(float x) {
  float x3 = x * x * x;
  return 0.5f * x * (1.0f + tanhf(0.7978845608028654f * (x + 0.044715f * x3)));
}

// ---------------- f32 -> f16 conversion ----------------
__global__ void cvt_f32_f16(const float* __restrict__ in, _Float16* __restrict__ out, int n) {
  int i = blockIdx.x * blockDim.x + threadIdx.x;
  int stride = gridDim.x * blockDim.x;
  for (; i < n; i += stride) out[i] = (_Float16)in[i];
}

// ---------------- LayerNorm (f32 in, f16 out), one row per block ----------------
__global__ __launch_bounds__(256) void layernorm_f16(
    const float* __restrict__ x, const float* __restrict__ g, const float* __restrict__ bta,
    _Float16* __restrict__ out, int D) {
  int row = blockIdx.x;
  const float* xr = x + (size_t)row * D;
  int lane = threadIdx.x & 31, wave = threadIdx.x >> 5;

  __shared__ float red[8];
  __shared__ float mu_s, rs_s;

  float s = 0.f;
  for (int i = threadIdx.x; i < D; i += 256) s += xr[i];
  #pragma unroll
  for (int o = 16; o > 0; o >>= 1) s += __shfl_xor(s, o, 32);
  if (lane == 0) red[wave] = s;
  __syncthreads();
  if (threadIdx.x == 0) {
    float t = 0.f;
    #pragma unroll
    for (int i = 0; i < 8; i++) t += red[i];
    mu_s = t / (float)D;
  }
  __syncthreads();
  float mu = mu_s;

  float v = 0.f;
  for (int i = threadIdx.x; i < D; i += 256) { float d = xr[i] - mu; v += d * d; }
  #pragma unroll
  for (int o = 16; o > 0; o >>= 1) v += __shfl_xor(v, o, 32);
  if (lane == 0) red[wave] = v;
  __syncthreads();
  if (threadIdx.x == 0) {
    float t = 0.f;
    #pragma unroll
    for (int i = 0; i < 8; i++) t += red[i];
    rs_s = rsqrtf(t / (float)D + 1e-5f);
  }
  __syncthreads();
  float rstd = rs_s;

  for (int i = threadIdx.x; i < D; i += 256) {
    float y = (xr[i] - mu) * rstd * g[i] + bta[i];
    out[(size_t)row * D + i] = (_Float16)y;
  }
}

// ---------------- generic WMMA GEMM: C = A(f16)[M,K] x B(f16)[K,N] + bias ----------------
// 256 threads = 8 waves (2x4), block tile 64x128, each wave 32x32 (2x2 WMMA tiles), K-step 32.
// A tile kept row-major (fragment = two contiguous 8-half runs).
// B tile stored TRANSPOSED (col-major over k) so B fragments are contiguous b128 pairs.
#define TM 64
#define TN 128
#define TK 32
#define ASTR 40    // padded LDS row stride (halves) for A tile rows
#define BTSTR 40   // padded LDS stride (halves) per B column (k-major)

__global__ __launch_bounds__(256) void wmma_gemm(
    const _Float16* __restrict__ A, const _Float16* __restrict__ Bm,
    const float* __restrict__ bias, int M, int N, int K,
    const float* __restrict__ resid,  // optional residual [M,N] f32
    float* __restrict__ outF,         // optional f32 out
    _Float16* __restrict__ outH,      // optional f16 out
    int doGelu) {
  __shared__ _Float16 As[TM * ASTR];        // 64 rows x 32 k (row-major)
  __shared__ _Float16 BsT[TN * BTSTR];      // 128 cols x 32 k (k contiguous per col)

  int tid = threadIdx.x;
  int wave = tid >> 5, lane = tid & 31;
  int wm = wave >> 2, wn = wave & 3;   // 2 x 4 wave grid
  int laneHalf = lane & 15;
  int hiSel = lane >> 4;               // 0 or 1

  int mBase = blockIdx.y * TM;
  int nBase = blockIdx.x * TN;

  int aRow = tid >> 2;            // 0..63
  int aCol = (tid & 3) * 8;       // 0..24
  int bColGrp = (tid & 15) * 8;   // 0..120 (8 columns)
  int bK2 = (tid >> 4) * 2;       // 0..30  (2 consecutive k rows)

  v8f acc[2][2] = {};

  for (int k0 = 0; k0 < K; k0 += TK) {
    // ---- A tile: coalesced b128 load, row-major LDS store ----
    *(H8*)&As[aRow * ASTR + aCol] =
        *(const H8*)&A[(size_t)(mBase + aRow) * K + k0 + aCol];

    // ---- B tile: 2 coalesced b128 row loads, transposed b32 LDS stores ----
    H8 r0 = *(const H8*)&Bm[(size_t)(k0 + bK2) * N + nBase + bColGrp];
    H8 r1 = *(const H8*)&Bm[(size_t)(k0 + bK2 + 1) * N + nBase + bColGrp];
    #pragma unroll
    for (int i = 0; i < 8; i++) {
      H2 t2{r0.h[i], r1.h[i]};
      *(H2*)&BsT[(bColGrp + i) * BTSTR + bK2] = t2;
    }

    if (k0 + TK < K) {
      __builtin_prefetch(&A[(size_t)(mBase + aRow) * K + k0 + TK + aCol], 0, 1);
    }
    __syncthreads();

    // ---- fragments: contiguous b128 pairs + free register concat ----
    v16h afrag[2], bfrag[2];
    #pragma unroll
    for (int mi = 0; mi < 2; mi++) {
      int row = wm * 32 + mi * 16 + laneHalf;
      v8h lo = *(const v8h*)&As[row * ASTR + hiSel * 8];
      v8h hi = *(const v8h*)&As[row * ASTR + hiSel * 8 + 16];
      afrag[mi] = cat16(lo, hi);
    }
    #pragma unroll
    for (int ni = 0; ni < 2; ni++) {
      int col = wn * 32 + ni * 16 + laneHalf;
      v8h lo = *(const v8h*)&BsT[col * BTSTR + hiSel * 8];
      v8h hi = *(const v8h*)&BsT[col * BTSTR + hiSel * 8 + 16];
      bfrag[ni] = cat16(lo, hi);
    }

    #pragma unroll
    for (int mi = 0; mi < 2; mi++)
      #pragma unroll
      for (int ni = 0; ni < 2; ni++)
        acc[mi][ni] = WMMA_F32_F16(afrag[mi], bfrag[ni], acc[mi][ni]);

    __syncthreads();
  }

  // ---- epilogue: bias (+gelu) (+residual), dual-precision store ----
  #pragma unroll
  for (int mi = 0; mi < 2; mi++) {
    #pragma unroll
    for (int ni = 0; ni < 2; ni++) {
      int col = nBase + wn * 32 + ni * 16 + laneHalf;
      float bv = bias ? bias[col] : 0.f;
      #pragma unroll
      for (int r = 0; r < 8; r++) {
        int row = mBase + wm * 32 + mi * 16 + hiSel * 8 + r;
        float v = acc[mi][ni][r] + bv;
        if (doGelu) v = gelu_new(v);
        size_t o = (size_t)row * N + col;
        if (resid) v += resid[o];
        if (outF) outF[o] = v;
        if (outH) outH[o] = (_Float16)v;
      }
    }
  }
}

// ---------------- causal flash attention (f16 in/out, f32 softmax state) ----------------
// qkv: [B*S, 3*D] f16 (q | k | v per row).  out: [B*S, D] f16 (heads re-merged).
// Block = 128 threads = 4 waves; each wave owns one 16-row query tile; streams 32-key blocks.
#define VSTR 40   // padded stride (halves) per hd column in the per-wave V^T buffer
#define PSTR 40   // padded stride (halves) per row in the per-wave P buffer

__global__ __launch_bounds__(128) void flash_attn(
    const _Float16* __restrict__ qkv, _Float16* __restrict__ outH) {
  __shared__ _Float16 pBuf[4][16 * PSTR];   // per-wave P staging (row-major, k contiguous)
  __shared__ _Float16 vT[4][HDd * VSTR];    // per-wave V^T (key contiguous per hd)

  int wave = threadIdx.x >> 5, lane = threadIdx.x & 31;
  int laneHalf = lane & 15, hiSel = lane >> 4;
  int h = blockIdx.y, b = blockIdx.z;
  int q0 = (blockIdx.x * 4 + wave) * 16;

  const int RS = 3 * Dd;  // qkv row stride (halves)
  const _Float16* base = qkv + (size_t)b * Ss * RS;
  const _Float16* qp = base + h * HDd;
  const _Float16* kp = base + Dd + h * HDd;
  const _Float16* vp = base + 2 * Dd + h * HDd;

  // Q fragments resident in registers (16 q-rows x 64 head dim = 2 K-chunks)
  v16h qf[2];
  {
    int row = q0 + laneHalf;
    #pragma unroll
    for (int kk = 0; kk < 2; kk++) {
      v8h lo = *(const v8h*)&qp[(size_t)row * RS + kk * 32 + hiSel * 8];
      v8h hi = *(const v8h*)&qp[(size_t)row * RS + kk * 32 + hiSel * 8 + 16];
      qf[kk] = cat16(lo, hi);
    }
  }

  v8f oacc[4] = {};
  float mrow[8], lrow[8];
  #pragma unroll
  for (int r = 0; r < 8; r++) { mrow[r] = -1e30f; lrow[r] = 0.f; }

  for (int kc0 = 0; kc0 <= q0 + 15; kc0 += 32) {
    // ---- scores: S = Q K^T for 16q x 32k (two 16x16 WMMA tiles, chained over hd) ----
    v8f s[2];
    #pragma unroll
    for (int g = 0; g < 2; g++) {
      int key = kc0 + g * 16 + laneHalf;
      v16h bf0, bf1;
      {
        v8h lo = *(const v8h*)&kp[(size_t)key * RS + hiSel * 8];
        v8h hi = *(const v8h*)&kp[(size_t)key * RS + hiSel * 8 + 16];
        bf0 = cat16(lo, hi);
      }
      {
        v8h lo = *(const v8h*)&kp[(size_t)key * RS + 32 + hiSel * 8];
        v8h hi = *(const v8h*)&kp[(size_t)key * RS + 32 + hiSel * 8 + 16];
        bf1 = cat16(lo, hi);
      }
      v8f z = {};
      z = WMMA_F32_F16(qf[0], bf0, z);
      z = WMMA_F32_F16(qf[1], bf1, z);
      s[g] = z;
    }

    // ---- stage V block (32 keys x 64 hd) transposed into per-wave LDS ----
    {
      int pair = laneHalf;              // keys 2*pair, 2*pair+1
      int hdBase = hiSel * 32;          // each half-wave covers 32 hd values
      const _Float16* vr0 = &vp[(size_t)(kc0 + 2 * pair) * RS + hdBase];
      const _Float16* vr1 = &vp[(size_t)(kc0 + 2 * pair + 1) * RS + hdBase];
      #pragma unroll
      for (int qq = 0; qq < 4; qq++) {
        v8h a = *(const v8h*)&vr0[qq * 8];
        v8h c = *(const v8h*)&vr1[qq * 8];
        #pragma unroll
        for (int i = 0; i < 8; i++) {
          H2 t2{a[i], c[i]};
          *(H2*)&vT[wave][(hdBase + qq * 8 + i) * VSTR + 2 * pair] = t2;
        }
      }
    }

    // ---- scale, causal mask, online softmax ----
    float p[2][8];
    #pragma unroll
    for (int r = 0; r < 8; r++) {
      int row = q0 + hiSel * 8 + r;
      #pragma unroll
      for (int g = 0; g < 2; g++) {
        int col = kc0 + g * 16 + laneHalf;
        float sv = s[g][r] * 0.125f;                // 1/sqrt(64)
        if (col > row) sv = -10000.0f;              // masked_bias, matches reference
        p[g][r] = sv;
      }
    }
    #pragma unroll
    for (int r = 0; r < 8; r++) {
      float mb = fmaxf(p[0][r], p[1][r]);
      #pragma unroll
      for (int o = 8; o > 0; o >>= 1) mb = fmaxf(mb, __shfl_xor(mb, o, 16));
      float mn = fmaxf(mrow[r], mb);
      float alpha = __expf(mrow[r] - mn);
      float e0 = __expf(p[0][r] - mn);
      float e1 = __expf(p[1][r] - mn);
      p[0][r] = e0; p[1][r] = e1;
      float rs = e0 + e1;
      #pragma unroll
      for (int o = 8; o > 0; o >>= 1) rs += __shfl_xor(rs, o, 16);
      lrow[r] = lrow[r] * alpha + rs;
      mrow[r] = mn;
      #pragma unroll
      for (int c = 0; c < 4; c++) oacc[c][r] *= alpha;
    }

    // ---- P: C-layout -> LDS -> A-fragment layout (same-wave LDS ops are in-order) ----
    #pragma unroll
    for (int g = 0; g < 2; g++)
      #pragma unroll
      for (int r = 0; r < 8; r++)
        pBuf[wave][(hiSel * 8 + r) * PSTR + g * 16 + laneHalf] = (_Float16)p[g][r];

    v16h pf;
    {
      v8h lo = *(const v8h*)&pBuf[wave][laneHalf * PSTR + hiSel * 8];
      v8h hi = *(const v8h*)&pBuf[wave][laneHalf * PSTR + hiSel * 8 + 16];
      pf = cat16(lo, hi);
    }

    // ---- O += P V  (4 hd chunks of 16), V fragments from transposed LDS ----
    #pragma unroll
    for (int c = 0; c < 4; c++) {
      int hd = c * 16 + laneHalf;
      v8h lo = *(const v8h*)&vT[wave][hd * VSTR + hiSel * 8];
      v8h hi = *(const v8h*)&vT[wave][hd * VSTR + hiSel * 8 + 16];
      v16h vfrag = cat16(lo, hi);
      oacc[c] = WMMA_F32_F16(pf, vfrag, oacc[c]);
    }
  }

  // ---- normalize and write merged-head output ----
  #pragma unroll
  for (int c = 0; c < 4; c++) {
    #pragma unroll
    for (int r = 0; r < 8; r++) {
      int row = q0 + hiSel * 8 + r;
      int col = h * HDd + c * 16 + laneHalf;
      float v = oacc[c][r] / lrow[r];
      outH[((size_t)b * Ss + row) * Dd + col] = (_Float16)v;
    }
  }
}

// ---------------- host-side orchestration ----------------
static inline char* ws_take(char*& p, size_t bytes) {
  char* r = p;
  p += (bytes + 255) & ~(size_t)255;
  return r;
}

extern "C" void kernel_launch(void* const* d_in, const int* in_sizes, int n_in,
                              void* d_out, int out_size, void* d_ws, size_t ws_size,
                              hipStream_t stream) {
  (void)in_sizes; (void)n_in; (void)out_size; (void)ws_size;

  const float* x      = (const float*)d_in[0];
  const float* ln1_g  = (const float*)d_in[1];
  const float* ln1_b  = (const float*)d_in[2];
  const float* w_attn = (const float*)d_in[3];
  const float* b_attn = (const float*)d_in[4];
  const float* w_o    = (const float*)d_in[5];
  const float* b_o    = (const float*)d_in[6];
  const float* ln2_g  = (const float*)d_in[7];
  const float* ln2_b  = (const float*)d_in[8];
  const float* w_fc   = (const float*)d_in[9];
  const float* b_fc   = (const float*)d_in[10];
  const float* w_fp   = (const float*)d_in[11];
  const float* b_fp   = (const float*)d_in[12];
  float* out = (float*)d_out;

  // workspace carve-up (~120 MB)
  char* p = (char*)d_ws;
  _Float16* h1    = (_Float16*)ws_take(p, (size_t)Mrows * Dd * 2);        // LN1(x) f16
  _Float16* wattnH= (_Float16*)ws_take(p, (size_t)Dd * 3 * Dd * 2);
  _Float16* qkvH  = (_Float16*)ws_take(p, (size_t)Mrows * 3 * Dd * 2);
  _Float16* attnH = (_Float16*)ws_take(p, (size_t)Mrows * Dd * 2);
  _Float16* woH   = (_Float16*)ws_take(p, (size_t)Dd * Dd * 2);
  float*    x1    = (float*)   ws_take(p, (size_t)Mrows * Dd * 4);        // attn residual out
  _Float16* h2    = (_Float16*)ws_take(p, (size_t)Mrows * Dd * 2);        // LN2(x1) f16
  _Float16* wfcH  = (_Float16*)ws_take(p, (size_t)Dd * 4 * Dd * 2);
  _Float16* fcH   = (_Float16*)ws_take(p, (size_t)Mrows * 4 * Dd * 2);    // gelu(fc) f16
  _Float16* wfpH  = (_Float16*)ws_take(p, (size_t)4 * Dd * Dd * 2);

  // 1) weight conversions f32 -> f16
  {
    int n;
    n = Dd * 3 * Dd; cvt_f32_f16<<<1024, 256, 0, stream>>>(w_attn, wattnH, n);
    n = Dd * Dd;     cvt_f32_f16<<<1024, 256, 0, stream>>>(w_o,    woH,    n);
    n = Dd * 4 * Dd; cvt_f32_f16<<<1024, 256, 0, stream>>>(w_fc,   wfcH,   n);
    n = 4 * Dd * Dd; cvt_f32_f16<<<1024, 256, 0, stream>>>(w_fp,   wfpH,   n);
  }

  // 2) LN1 -> h1 (f16)
  layernorm_f16<<<Mrows, 256, 0, stream>>>(x, ln1_g, ln1_b, h1, Dd);

  // 3) QKV = h1 @ Wattn + b_attn  -> qkvH (f16)
  wmma_gemm<<<dim3(3 * Dd / TN, Mrows / TM), 256, 0, stream>>>(
      h1, wattnH, b_attn, Mrows, 3 * Dd, Dd, nullptr, nullptr, qkvH, 0);

  // 4) causal flash attention -> attnH (f16, heads merged)
  flash_attn<<<dim3(Ss / 64, Hh, Bb), 128, 0, stream>>>(qkvH, attnH);

  // 5) x1 = attnH @ Wo + b_o + x  (f32)
  wmma_gemm<<<dim3(Dd / TN, Mrows / TM), 256, 0, stream>>>(
      attnH, woH, b_o, Mrows, Dd, Dd, x, x1, nullptr, 0);

  // 6) LN2 -> h2 (f16)
  layernorm_f16<<<Mrows, 256, 0, stream>>>(x1, ln2_g, ln2_b, h2, Dd);

  // 7) fc = gelu(h2 @ Wfc + b_fc) -> fcH (f16)
  wmma_gemm<<<dim3(4 * Dd / TN, Mrows / TM), 256, 0, stream>>>(
      h2, wfcH, b_fc, Mrows, 4 * Dd, Dd, nullptr, nullptr, fcH, 1);

  // 8) out = fcH @ Wfp + b_fp + x1  (f32, final)
  wmma_gemm<<<dim3(Dd / TN, Mrows / TM), 256, 0, stream>>>(
      fcH, wfpH, b_fp, Mrows, Dd, 4 * Dd, x1, out, nullptr, 0);
}
